// GNN_v1_30932354465859
// MI455X (gfx1250) — matrix operand order repaired and checked by reading, not compiled, not run on hardware
//
#include <hip/hip_runtime.h>
#include <hip/hip_bf16.h>

typedef __attribute__((ext_vector_type(16))) _Float16 v16h;
typedef __attribute__((ext_vector_type(8)))  float    v8f;

#define N_NODES 100000
#define IN_CH   256
#define OUT_CH  64
#define N_EDGES 3200000
#define M_TILES (N_NODES / 16)   // 6250 exactly

// ---------------------------------------------------------------------------
// Kernel 1: init accum = 0, deg = 1 (self loop contribution)
// ---------------------------------------------------------------------------
__global__ void k_init(float* __restrict__ accum, float* __restrict__ deg) {
    size_t t = (size_t)blockIdx.x * blockDim.x + threadIdx.x;
    if (t < (size_t)N_NODES * OUT_CH) accum[t] = 0.0f;
    if (t < (size_t)N_NODES)          deg[t]   = 1.0f;
}

// ---------------------------------------------------------------------------
// Kernel 2: degree accumulation over edges (dst side)
// ---------------------------------------------------------------------------
__global__ void k_degree(const long long* __restrict__ dst, float* __restrict__ deg) {
    size_t e = (size_t)blockIdx.x * blockDim.x + threadIdx.x;
    if (e < (size_t)N_EDGES) {
        atomicAdd(&deg[(int)dst[e]], 1.0f);
    }
}

// ---------------------------------------------------------------------------
// Kernel 3: dinv = rsqrt(deg) in place (deg >= 1 always due to self loop)
// ---------------------------------------------------------------------------
__global__ void k_dinv(float* __restrict__ deg) {
    size_t i = (size_t)blockIdx.x * blockDim.x + threadIdx.x;
    if (i < (size_t)N_NODES) deg[i] = rsqrtf(deg[i]);
}

// ---------------------------------------------------------------------------
// Kernel 4: h = x @ W via v_wmma_f32_16x16x32_f16
//   - 256 threads = 8 waves per block, 1 row-tile (16 rows x 64 cols) per wave
//   - W transposed to f16 in LDS: Wt[col][k], contiguous K per column
// ---------------------------------------------------------------------------
__global__ void __launch_bounds__(256) k_gemm(const float* __restrict__ x,
                                              const float* __restrict__ W,
                                              float* __restrict__ h) {
    __shared__ _Float16 Wt[OUT_CH * IN_CH];   // 32 KB, [col][k]

    // cooperative transpose+convert of W (256x64 f32 -> 64x256 f16)
    for (int i = threadIdx.x; i < OUT_CH * IN_CH; i += 256) {
        int col = i >> 8;          // 0..63
        int k   = i & 255;         // 0..255
        Wt[i] = (_Float16)W[k * OUT_CH + col];
    }
    __syncthreads();

    const int wave = threadIdx.x >> 5;
    const int lane = threadIdx.x & 31;
    const int tile = blockIdx.x * 8 + wave;
    if (tile >= M_TILES) return;

    const int hi  = lane >> 4;            // 0 = lanes 0-15, 1 = lanes 16-31
    const int row = tile * 16 + (lane & 15);
    const float* xrow = x + (size_t)row * IN_CH + hi * 16;  // A K-offset: 0 or 16

    v8f acc0 = {}, acc1 = {}, acc2 = {}, acc3 = {};

    for (int k0 = 0; k0 < IN_CH; k0 += 32) {
        // --- A fragment (16-bit A 16x32, contiguous K):
        //   lanes 0-15 : elems 0-15 = K k0+0..15
        //   lanes 16-31: elems 0-15 = K k0+16..31
        float4 f0 = *(const float4*)(xrow + k0 + 0);
        float4 f1 = *(const float4*)(xrow + k0 + 4);
        float4 f2 = *(const float4*)(xrow + k0 + 8);
        float4 f3 = *(const float4*)(xrow + k0 + 12);
        v16h a;
        a[0]  = (_Float16)f0.x; a[1]  = (_Float16)f0.y;
        a[2]  = (_Float16)f0.z; a[3]  = (_Float16)f0.w;
        a[4]  = (_Float16)f1.x; a[5]  = (_Float16)f1.y;
        a[6]  = (_Float16)f1.z; a[7]  = (_Float16)f1.w;
        a[8]  = (_Float16)f2.x; a[9]  = (_Float16)f2.y;
        a[10] = (_Float16)f2.z; a[11] = (_Float16)f2.w;
        a[12] = (_Float16)f3.x; a[13] = (_Float16)f3.y;
        a[14] = (_Float16)f3.z; a[15] = (_Float16)f3.w;

        // --- B fragments (32x16, 16-bit): lanes 0-15 hold K k0..k0+15 of
        // column N=lane; lanes 16-31 hold K k0+16..k0+31 of column N=lane-16.
        const int kb = k0 + hi * 16;
        const int nc = lane & 15;
        v16h b0 = *(const v16h*)&Wt[(nc +  0) * IN_CH + kb];
        v16h b1 = *(const v16h*)&Wt[(nc + 16) * IN_CH + kb];
        v16h b2 = *(const v16h*)&Wt[(nc + 32) * IN_CH + kb];
        v16h b3 = *(const v16h*)&Wt[(nc + 48) * IN_CH + kb];

        acc0 = __builtin_amdgcn_wmma_f32_16x16x32_f16(false, a, false, b0,
                                                      (short)0, acc0, false, false);
        acc1 = __builtin_amdgcn_wmma_f32_16x16x32_f16(false, a, false, b1,
                                                      (short)0, acc1, false, false);
        acc2 = __builtin_amdgcn_wmma_f32_16x16x32_f16(false, a, false, b2,
                                                      (short)0, acc2, false, false);
        acc3 = __builtin_amdgcn_wmma_f32_16x16x32_f16(false, a, false, b3,
                                                      (short)0, acc3, false, false);
    }

    // --- store C/D (f32 16x16): VGPR r -> M = r + hi*8, N = lane&15
    const int m0 = tile * 16 + hi * 8;
    const int nc = lane & 15;
    float* hp = h + (size_t)m0 * OUT_CH + nc;
#pragma unroll
    for (int r = 0; r < 8; ++r) {
        hp[(size_t)r * OUT_CH +  0] = acc0[r];
        hp[(size_t)r * OUT_CH + 16] = acc1[r];
        hp[(size_t)r * OUT_CH + 32] = acc2[r];
        hp[(size_t)r * OUT_CH + 48] = acc3[r];
    }
}

// ---------------------------------------------------------------------------
// Kernel 5: edge scatter. 16 threads per edge, float4 gather + 4 f32 atomics.
// h and accum are L2-resident (25.6 MB each, 192 MB L2).
// ---------------------------------------------------------------------------
__global__ void k_scatter(const long long* __restrict__ src,
                          const long long* __restrict__ dst,
                          const float* __restrict__ h,
                          const float* __restrict__ dinv,
                          float* __restrict__ accum) {
    size_t t = (size_t)blockIdx.x * blockDim.x + threadIdx.x;
    size_t e = t >> 4;
    if (e >= (size_t)N_EDGES) return;
    int part = (int)(t & 15);

    int s = (int)src[e];
    int d = (int)dst[e];
    float nrm = dinv[s] * dinv[d];

    float4 hv = *(const float4*)(h + (size_t)s * OUT_CH + part * 4);
    float* op = accum + (size_t)d * OUT_CH + part * 4;
    atomicAdd(op + 0, hv.x * nrm);
    atomicAdd(op + 1, hv.y * nrm);
    atomicAdd(op + 2, hv.z * nrm);
    atomicAdd(op + 3, hv.w * nrm);
}

// ---------------------------------------------------------------------------
// Kernel 6: self-loop term + bias + relu + log_softmax. One wave32 per node,
// 2 channels per lane, wave-wide shuffle reductions.
// ---------------------------------------------------------------------------
__global__ void k_final(const float* __restrict__ accum,
                        const float* __restrict__ h,
                        const float* __restrict__ dinv,
                        const float* __restrict__ bias,
                        float* __restrict__ out) {
    size_t t = (size_t)blockIdx.x * blockDim.x + threadIdx.x;
    int node = (int)(t >> 5);
    if (node >= N_NODES) return;
    int lane = (int)(t & 31);
    int c = lane * 2;

    float di = dinv[node];
    float selfw = di * di;
    size_t base = (size_t)node * OUT_CH + c;

    float v0 = accum[base + 0] + h[base + 0] * selfw + bias[c + 0];
    float v1 = accum[base + 1] + h[base + 1] * selfw + bias[c + 1];
    v0 = fmaxf(v0, 0.0f);
    v1 = fmaxf(v1, 0.0f);

    float m = fmaxf(v0, v1);
#pragma unroll
    for (int off = 16; off >= 1; off >>= 1)
        m = fmaxf(m, __shfl_xor(m, off, 32));

    float s = expf(v0 - m) + expf(v1 - m);
#pragma unroll
    for (int off = 16; off >= 1; off >>= 1)
        s += __shfl_xor(s, off, 32);

    float lse = logf(s);
    out[base + 0] = v0 - m - lse;
    out[base + 1] = v1 - m - lse;
}

// ---------------------------------------------------------------------------
extern "C" void kernel_launch(void* const* d_in, const int* in_sizes, int n_in,
                              void* d_out, int out_size, void* d_ws, size_t ws_size,
                              hipStream_t stream) {
    const float*     x  = (const float*)d_in[0];      // [N, 256] f32
    const long long* ei = (const long long*)d_in[1];  // [2, E]  int64
    const float*     W  = (const float*)d_in[2];      // [256, 64] f32
    const float*     b  = (const float*)d_in[3];      // [64] f32
    float*           out = (float*)d_out;             // [N, 64] f32

    const long long* src = ei;
    const long long* dst = ei + N_EDGES;

    // workspace layout: h | accum | deg(->dinv)
    float* hbuf  = (float*)d_ws;
    float* accum = hbuf  + (size_t)N_NODES * OUT_CH;
    float* degv  = accum + (size_t)N_NODES * OUT_CH;

    const int TB = 256;

    // 1) accum = 0, deg = 1
    {
        size_t n = (size_t)N_NODES * OUT_CH;
        k_init<<<(unsigned)((n + TB - 1) / TB), TB, 0, stream>>>(accum, degv);
    }
    // 2) degree atomics
    k_degree<<<(N_EDGES + TB - 1) / TB, TB, 0, stream>>>(dst, degv);
    // 3) dinv = rsqrt(deg)
    k_dinv<<<(N_NODES + TB - 1) / TB, TB, 0, stream>>>(degv);
    // 4) h = x @ W (WMMA f16->f32)
    k_gemm<<<(M_TILES + 7) / 8, TB, 0, stream>>>(x, W, hbuf);
    // 5) edge scatter-add
    {
        size_t n = (size_t)N_EDGES * 16;
        k_scatter<<<(unsigned)((n + TB - 1) / TB), TB, 0, stream>>>(src, dst, hbuf, degv, accum);
    }
    // 6) finalize: relu + log_softmax
    {
        size_t n = (size_t)N_NODES * 32;
        k_final<<<(unsigned)((n + TB - 1) / TB), TB, 0, stream>>>(accum, hbuf, degv, b, out);
    }
}
